// TemporalEncoder_87119116632104
// MI455X (gfx1250) — matrix operand order repaired and checked by reading, not compiled, not run on hardware
//
#include <hip/hip_runtime.h>
#include <cstdint>
#include <cstddef>

#define H_    64
#define FEAT_ 128
#define EMB_  128
#define B_    512
#define T_    512
#define G4H   256   // 4*H

typedef __attribute__((ext_vector_type(16))) __bf16 v16bf;
typedef __attribute__((ext_vector_type(2)))  __bf16 v2bf;
typedef __attribute__((ext_vector_type(8)))  float  v8f;

union Frag {
    v16bf v;
    uint4 q[2];
    unsigned short s[16];
};

#if defined(__has_builtin)
#if __has_builtin(__builtin_amdgcn_cvt_pk_bf16_f32)
#define HAVE_PK_BF16 1
#endif
#if __has_builtin(__builtin_amdgcn_tanhf)
#define HAVE_NATIVE_TANH 1
#endif
#endif

__device__ __forceinline__ unsigned int pack2(float a, float b) {
#ifdef HAVE_PK_BF16
    union { v2bf v; unsigned int u; } cv;
    cv.v = __builtin_amdgcn_cvt_pk_bf16_f32(a, b);
    return cv.u;
#else
    unsigned int r;
    asm("v_cvt_pk_bf16_f32 %0, %1, %2" : "=v"(r) : "v"(a), "v"(b));
    return r;
#endif
}

__device__ __forceinline__ unsigned short f2bf(float f) {
    return (unsigned short)(pack2(f, 0.0f) & 0xffffu);
}

__device__ __forceinline__ float fast_tanh(float x) {
#ifdef HAVE_NATIVE_TANH
    return __builtin_amdgcn_tanhf(x);
#else
    float y;
    // TRANS op: guarantee one slot before the result is consumed (ISA 7.4)
    asm("v_tanh_f32 %0, %1\n\tv_nop" : "=v"(y) : "v"(x));
    return y;
#endif
}
__device__ __forceinline__ float fast_sig(float x) {
    return 0.5f + 0.5f * fast_tanh(0.5f * x);
}

// xp layout: [dir][chunk][t][wave*4+gate][lane] -> 16 bytes (8 bf16, C-frag rows)
__device__ __forceinline__ size_t xp_off(int dir, int chunk, int t, int wg, int l) {
    return ((((((size_t)dir * 32 + chunk) * T_ + t) * 16 + wg) * 32) + l) * 16;
}

// ---------------------------------------------------------------------------
// Phase A: xp = x @ W + b  (bf16, stored in WMMA C-fragment layout)
// Grid: 1024 blocks (2 dir x 32 chunks x 16 t-slices) x 128 threads.
// ---------------------------------------------------------------------------
__global__ void __launch_bounds__(128) xp_gemm_kernel(
    const float* __restrict__ x,
    const float* __restrict__ Wf, const float* __restrict__ bf_,
    const float* __restrict__ Wb, const float* __restrict__ bb_,
    unsigned char* __restrict__ xp) {

    __shared__ __align__(16) unsigned short Wl[48 * 512];        // 48 KB
    __shared__ __align__(16) unsigned char xa_base[2][4 * 1024]; //  8 KB

    const int tid = threadIdx.x;
    const int l   = tid & 31;
    const int w   = tid >> 5;
    const int dir    = blockIdx.x & 1;
    const int chunk  = (blockIdx.x >> 1) & 31;
    const int tslice = blockIdx.x >> 6;
    const int batch_base = chunk * 16;

    const float* W    = dir ? Wb  : Wf;
    const float* bias = dir ? bb_ : bf_;

    for (int idx = tid; idx < 48 * 512; idx += 128) {
        int e  = idx & 15;
        int ln = (idx >> 4) & 31;
        int f  = idx >> 9;            // ((w*4+g)*3 + kb)
        int kb = f % 3;
        int g  = (f / 3) & 3;
        int wv = f / 12;
        int K  = kb * 32 + (ln >> 4) * 16 + e;
        int N  = wv * 16 + g * 64 + (ln & 15);
        Wl[idx] = f2bf(W[K * G4H + N]);
    }
    Frag wreg[4];
    float bgate[4];
    for (int g = 0; g < 4; ++g) {
        int N = w * 16 + g * 64 + (l & 15);
        for (int e = 0; e < 16; ++e) {
            int K3 = 96 + (l >> 4) * 16 + e;
            wreg[g].s[e] = f2bf(W[K3 * G4H + N]);
        }
        bgate[g] = bias[g * 64 + w * 16 + (l & 15)];
    }

    for (int ti = 0; ti < 32; ++ti) {
        int t = tslice * 32 + ti;
        int t_eff = dir ? (T_ - 1 - t) : t;
        unsigned char* xab = xa_base[ti & 1];
#pragma unroll
        for (int rr = 0; rr < 2; ++rr) {
            int run = tid * 2 + rr;
            int row = run >> 4;
            int K0  = (run & 15) * 8;
            const float* src =
                x + ((size_t)(batch_base + row) * T_ + t_eff) * FEAT_ + K0;
            float4 a0 = *(const float4*)src;
            float4 a1 = *(const float4*)(src + 4);
            uint4 p;
            p.x = pack2(a0.x, a0.y);
            p.y = pack2(a0.z, a0.w);
            p.z = pack2(a1.x, a1.y);
            p.w = pack2(a1.z, a1.w);
            int kb   = K0 >> 5;
            int Kp   = K0 & 31;
            int lane = ((Kp >> 4) << 4) + row;
            int e0   = Kp & 15;
            *(uint4*)(xab + kb * 1024 + lane * 32 + e0 * 2) = p;
        }
        __syncthreads();

        Frag xa[4];
#pragma unroll
        for (int kb = 0; kb < 4; ++kb) {
            const uint4* p = (const uint4*)(xab + kb * 1024 + l * 32);
            xa[kb].q[0] = p[0];
            xa[kb].q[1] = p[1];
        }
#pragma unroll
        for (int g = 0; g < 4; ++g) {
            v8f a;
#pragma unroll
            for (int r = 0; r < 8; ++r) a[r] = bgate[g];
#pragma unroll
            for (int kb = 0; kb < 3; ++kb) {
                Frag wf_;
                const uint4* p = (const uint4*)((const unsigned char*)Wl +
                                  ((w * 4 + g) * 3 + kb) * 1024 + l * 32);
                wf_.q[0] = p[0];
                wf_.q[1] = p[1];
                a = __builtin_amdgcn_wmma_f32_16x16x32_bf16(
                        false, xa[kb].v, false, wf_.v, (short)0, a, false, false);
            }
            a = __builtin_amdgcn_wmma_f32_16x16x32_bf16(
                    false, xa[3].v, false, wreg[g].v, (short)0, a, false, false);
            uint4 p;
            p.x = pack2(a[0], a[1]);
            p.y = pack2(a[2], a[3]);
            p.z = pack2(a[4], a[5]);
            p.w = pack2(a[6], a[7]);
            *(uint4*)(xp + xp_off(dir, chunk, t, w * 4 + g, l)) = p;
        }
    }
}

// ---------------------------------------------------------------------------
// Phase B: sequential scan reading precomputed xp; only h@U on critical path.
// Grid: 64 blocks x 128 threads. 8 v_wmma + ONE barrier per step.
// ---------------------------------------------------------------------------
__global__ void __launch_bounds__(128) lstm_scan_xp_kernel(
    const unsigned char* __restrict__ xp,
    const float* __restrict__ Uf, const float* __restrict__ Ub,
    float* __restrict__ h_all) {

    __shared__ __align__(16) unsigned short hl[2][16 * 64];      // 4 KB

    const int tid = threadIdx.x;
    const int l   = tid & 31;
    const int w   = tid >> 5;
    const int dir = blockIdx.x & 1;
    const int chunk = blockIdx.x >> 1;
    const int batch_base = chunk * 16;

    const float* U = dir ? Ub : Uf;

    Frag uf[4][2];
    for (int g = 0; g < 4; ++g)
        for (int kb = 0; kb < 2; ++kb)
            for (int e = 0; e < 16; ++e) {
                int K = kb * 32 + (l >> 4) * 16 + e;
                int N = w * 16 + g * 64 + (l & 15);
                uf[g][kb].s[e] = f2bf(U[K * G4H + N]);
            }

    float creg[8], hreg[8];
#pragma unroll
    for (int r = 0; r < 8; ++r) { creg[r] = 0.0f; hreg[r] = 0.0f; }

    const int hcol  = w * 16 + (l & 15);
    const int hrow0 = (l >> 4) * 8;

    uint4 xpf[4];
#pragma unroll
    for (int g = 0; g < 4; ++g)
        xpf[g] = *(const uint4*)(xp + xp_off(dir, chunk, 0, w * 4 + g, l));

    for (int t = 0; t < T_; ++t) {
        unsigned short* hb = hl[t & 1];
#pragma unroll
        for (int r = 0; r < 8; ++r)
            hb[(hrow0 + r) * 64 + hcol] = f2bf(hreg[r]);
        __syncthreads();

        Frag ha[2];
#pragma unroll
        for (int kb = 0; kb < 2; ++kb) {
#pragma unroll
            for (int j = 0; j < 2; ++j) {
                int K0 = kb * 32 + (l >> 4) * 16 + j * 8;
                ha[kb].q[j] = *(const uint4*)(hb + (l & 15) * 64 + K0);
            }
        }

        // issue next step's xp loads early; consumed next iteration
        uint4 xpn[4];
#pragma unroll
        for (int g = 0; g < 4; ++g) xpn[g] = make_uint4(0u, 0u, 0u, 0u);
        if (t + 1 < T_) {
#pragma unroll
            for (int g = 0; g < 4; ++g)
                xpn[g] = *(const uint4*)(xp + xp_off(dir, chunk, t + 1, w * 4 + g, l));
            if (t + 2 < T_)
                __builtin_prefetch(xp + xp_off(dir, chunk, t + 2, w * 4, l), 0, 3);
        }

        v8f acc[4];
#pragma unroll
        for (int g = 0; g < 4; ++g) {
            acc[g][0] = __uint_as_float(xpf[g].x << 16);
            acc[g][1] = __uint_as_float(xpf[g].x & 0xffff0000u);
            acc[g][2] = __uint_as_float(xpf[g].y << 16);
            acc[g][3] = __uint_as_float(xpf[g].y & 0xffff0000u);
            acc[g][4] = __uint_as_float(xpf[g].z << 16);
            acc[g][5] = __uint_as_float(xpf[g].z & 0xffff0000u);
            acc[g][6] = __uint_as_float(xpf[g].w << 16);
            acc[g][7] = __uint_as_float(xpf[g].w & 0xffff0000u);
#pragma unroll
            for (int kb = 0; kb < 2; ++kb)
                acc[g] = __builtin_amdgcn_wmma_f32_16x16x32_bf16(
                             false, ha[kb].v, false, uf[g][kb].v, (short)0,
                             acc[g], false, false);
        }

#pragma unroll
        for (int r = 0; r < 8; ++r) {
            float ig = fast_sig(acc[0][r]);
            float fg = fast_sig(acc[1][r]);
            float gg = fast_tanh(acc[2][r]);
            float og = fast_sig(acc[3][r]);
            creg[r] = fg * creg[r] + ig * gg;
            hreg[r] = og * fast_tanh(creg[r]);
        }
#pragma unroll
        for (int g = 0; g < 4; ++g) xpf[g] = xpn[g];
    }

#pragma unroll
    for (int r = 0; r < 8; ++r)
        h_all[(batch_base + hrow0 + r) * (2 * H_) + dir * 64 + hcol] = hreg[r];
}

// ---------------------------------------------------------------------------
// Fallback: fused scan, used when the workspace is too small for xp.
// ---------------------------------------------------------------------------
__global__ void __launch_bounds__(128) lstm_scan_fused_kernel(
    const float* __restrict__ x,
    const float* __restrict__ Wf, const float* __restrict__ Uf, const float* __restrict__ bf_,
    const float* __restrict__ Wb, const float* __restrict__ Ub, const float* __restrict__ bb_,
    float* __restrict__ h_all) {

    __shared__ __align__(16) unsigned short Wl[48 * 512];        // 48 KB
    __shared__ __align__(16) unsigned char xa_base[2][4 * 1024]; //  8 KB
    __shared__ __align__(16) unsigned short hl[2][16 * 64];      //  4 KB

    const int tid = threadIdx.x;
    const int l   = tid & 31;
    const int w   = tid >> 5;
    const int dir = blockIdx.x & 1;
    const int batch_base = (blockIdx.x >> 1) * 16;

    const float* W    = dir ? Wb  : Wf;
    const float* U    = dir ? Ub  : Uf;
    const float* bias = dir ? bb_ : bf_;

    for (int idx = tid; idx < 48 * 512; idx += 128) {
        int e  = idx & 15;
        int ln = (idx >> 4) & 31;
        int f  = idx >> 9;
        int kb = f % 3;
        int g  = (f / 3) & 3;
        int wv = f / 12;
        int K  = kb * 32 + (ln >> 4) * 16 + e;
        int N  = wv * 16 + g * 64 + (ln & 15);
        Wl[idx] = f2bf(W[K * G4H + N]);
    }
    Frag wreg[4];
    Frag uf[4][2];
    for (int g = 0; g < 4; ++g) {
        int N = w * 16 + g * 64 + (l & 15);
        for (int e = 0; e < 16; ++e) {
            int K3 = 96 + (l >> 4) * 16 + e;
            wreg[g].s[e] = f2bf(W[K3 * G4H + N]);
        }
        for (int kb = 0; kb < 2; ++kb)
            for (int e = 0; e < 16; ++e) {
                int K = kb * 32 + (l >> 4) * 16 + e;
                uf[g][kb].s[e] = f2bf(U[K * G4H + N]);
            }
    }
    float bgate[4];
    for (int g = 0; g < 4; ++g) bgate[g] = bias[g * 64 + w * 16 + (l & 15)];

    float creg[8], hreg[8];
#pragma unroll
    for (int r = 0; r < 8; ++r) { creg[r] = 0.0f; hreg[r] = 0.0f; }

    const int hcol  = w * 16 + (l & 15);
    const int hrow0 = (l >> 4) * 8;

    for (int t = 0; t < T_; ++t) {
        unsigned short* hb  = hl[t & 1];
        unsigned char*  xab = xa_base[t & 1];
#pragma unroll
        for (int r = 0; r < 8; ++r)
            hb[(hrow0 + r) * 64 + hcol] = f2bf(hreg[r]);

        const int t_eff = dir ? (T_ - 1 - t) : t;
#pragma unroll
        for (int rr = 0; rr < 2; ++rr) {
            int run = tid * 2 + rr;
            int row = run >> 4;
            int K0  = (run & 15) * 8;
            const float* src =
                x + ((size_t)(batch_base + row) * T_ + t_eff) * FEAT_ + K0;
            float4 a0 = *(const float4*)src;
            float4 a1 = *(const float4*)(src + 4);
            uint4 p;
            p.x = pack2(a0.x, a0.y);
            p.y = pack2(a0.z, a0.w);
            p.z = pack2(a1.x, a1.y);
            p.w = pack2(a1.z, a1.w);
            int kb   = K0 >> 5;
            int Kp   = K0 & 31;
            int lane = ((Kp >> 4) << 4) + row;
            int e0   = Kp & 15;
            *(uint4*)(xab + kb * 1024 + lane * 32 + e0 * 2) = p;
            if (t + 1 < T_) {
                int t_nx = dir ? (T_ - 2 - t) : (t + 1);
                __builtin_prefetch(
                    x + ((size_t)(batch_base + row) * T_ + t_nx) * FEAT_ + K0, 0, 3);
            }
        }
        __syncthreads();

        Frag xa[4];
#pragma unroll
        for (int kb = 0; kb < 4; ++kb) {
            const uint4* p = (const uint4*)(xab + kb * 1024 + l * 32);
            xa[kb].q[0] = p[0];
            xa[kb].q[1] = p[1];
        }
        Frag ha[2];
#pragma unroll
        for (int kb = 0; kb < 2; ++kb) {
#pragma unroll
            for (int j = 0; j < 2; ++j) {
                int K0 = kb * 32 + (l >> 4) * 16 + j * 8;
                ha[kb].q[j] = *(const uint4*)(hb + (l & 15) * 64 + K0);
            }
        }

        v8f acc[4];
#pragma unroll
        for (int g = 0; g < 4; ++g) {
            v8f a;
#pragma unroll
            for (int r = 0; r < 8; ++r) a[r] = bgate[g];
#pragma unroll
            for (int kb = 0; kb < 3; ++kb) {
                Frag wf_;
                const uint4* p = (const uint4*)((const unsigned char*)Wl +
                                  ((w * 4 + g) * 3 + kb) * 1024 + l * 32);
                wf_.q[0] = p[0];
                wf_.q[1] = p[1];
                a = __builtin_amdgcn_wmma_f32_16x16x32_bf16(
                        false, xa[kb].v, false, wf_.v, (short)0, a, false, false);
            }
            a = __builtin_amdgcn_wmma_f32_16x16x32_bf16(
                    false, xa[3].v, false, wreg[g].v, (short)0, a, false, false);
#pragma unroll
            for (int kb = 0; kb < 2; ++kb)
                a = __builtin_amdgcn_wmma_f32_16x16x32_bf16(
                        false, ha[kb].v, false, uf[g][kb].v, (short)0, a, false, false);
            acc[g] = a;
        }

#pragma unroll
        for (int r = 0; r < 8; ++r) {
            float ig = fast_sig(acc[0][r]);
            float fg = fast_sig(acc[1][r]);
            float gg = fast_tanh(acc[2][r]);
            float og = fast_sig(acc[3][r]);
            creg[r] = fg * creg[r] + ig * gg;
            hreg[r] = og * fast_tanh(creg[r]);
        }
    }

#pragma unroll
    for (int r = 0; r < 8; ++r)
        h_all[(batch_base + hrow0 + r) * (2 * H_) + dir * 64 + hcol] = hreg[r];
}

// ---------------------------------------------------------------------------
// Dense: out = relu(h_all @ W_dense + b_dense), [512,128]x[128,128]
// ---------------------------------------------------------------------------
__global__ void __launch_bounds__(128) dense_relu_kernel(
    const float* __restrict__ h_all, const float* __restrict__ Wd,
    const float* __restrict__ bd, float* __restrict__ out) {

    const int tid   = threadIdx.x;
    const int l     = tid & 31;
    const int w     = tid >> 5;
    const int m     = blockIdx.x;
    const int row_a = m * 16 + (l & 15);

#pragma unroll
    for (int nt2 = 0; nt2 < 2; ++nt2) {
        int nt  = w * 2 + nt2;
        int col = nt * 16 + (l & 15);
        v8f acc;
        float bv = bd[col];
#pragma unroll
        for (int r = 0; r < 8; ++r) acc[r] = bv;
#pragma unroll
        for (int kb = 0; kb < 4; ++kb) {
            Frag a;
#pragma unroll
            for (int j = 0; j < 2; ++j) {
                int K0 = kb * 32 + (l >> 4) * 16 + j * 8;
                const float* src = h_all + row_a * 128 + K0;
                float4 a0 = *(const float4*)src;
                float4 a1 = *(const float4*)(src + 4);
                a.q[j].x = pack2(a0.x, a0.y);
                a.q[j].y = pack2(a0.z, a0.w);
                a.q[j].z = pack2(a1.x, a1.y);
                a.q[j].w = pack2(a1.z, a1.w);
            }
            Frag bfr;
#pragma unroll
            for (int e = 0; e < 16; ++e) {
                int K = kb * 32 + (l >> 4) * 16 + e;
                bfr.s[e] = f2bf(Wd[K * EMB_ + col]);
            }
            acc = __builtin_amdgcn_wmma_f32_16x16x32_bf16(
                      false, a.v, false, bfr.v, (short)0, acc, false, false);
        }
#pragma unroll
        for (int r = 0; r < 8; ++r) {
            int row = m * 16 + (l >> 4) * 8 + r;
            float v = acc[r];
            out[row * EMB_ + col] = v > 0.0f ? v : 0.0f;
        }
    }
}

extern "C" void kernel_launch(void* const* d_in, const int* in_sizes, int n_in,
                              void* d_out, int out_size, void* d_ws, size_t ws_size,
                              hipStream_t stream) {
    (void)in_sizes; (void)n_in; (void)out_size;
    const float* x   = (const float*)d_in[0];
    const float* Wf  = (const float*)d_in[1];
    const float* Uf  = (const float*)d_in[2];
    const float* bf_ = (const float*)d_in[3];
    const float* Wb  = (const float*)d_in[4];
    const float* Ub  = (const float*)d_in[5];
    const float* bb_ = (const float*)d_in[6];
    const float* Wd  = (const float*)d_in[7];
    const float* bd  = (const float*)d_in[8];
    float* out   = (float*)d_out;
    float* h_all = (float*)d_ws;                       // 256 KB

    const size_t XP_BYTES = (size_t)1 << 28;           // 256 MB bf16 xp
    const size_t XP_BASE  = (size_t)1 << 20;           // after h_all, aligned

    if (ws_size >= XP_BASE + XP_BYTES) {
        unsigned char* xp = (unsigned char*)d_ws + XP_BASE;
        xp_gemm_kernel<<<dim3(1024), dim3(128), 0, stream>>>(
            x, Wf, bf_, Wb, bb_, xp);
        lstm_scan_xp_kernel<<<dim3(64), dim3(128), 0, stream>>>(
            xp, Uf, Ub, h_all);
    } else {
        lstm_scan_fused_kernel<<<dim3(64), dim3(128), 0, stream>>>(
            x, Wf, Uf, bf_, Wb, Ub, bb_, h_all);
    }
    dense_relu_kernel<<<dim3(32), dim3(128), 0, stream>>>(h_all, Wd, bd, out);
}
